// ResonanceLayer_67267777790351
// MI455X (gfx1250) — matrix operand order repaired and checked by reading, not compiled, not run on hardware
//
#include <hip/hip_runtime.h>
#include <math.h>

// ---------------- problem constants ----------------
#define NSAMP   131072
#define CPD     16
#define NRES    16
#define NCH     32            // NRES * EXPR
#define ATTACK  2048
#define FRAMES  128
#define NEV     4
#define EPSF    1e-8f

// ---------------- conv tiling constants ----------------
#define SEG     4096
#define NSEG    (NSAMP / SEG)            // 32
#define HIST    1568                     // max tap lookback 1567
#define RLEN    (SEG + HIST)             // 5664 (reversed signal in LDS)
#define REVPAD  132736                   // NSAMP + HIST padded so 2*REVPAD % 256 == 0
#define ACCLEN  (SEG + 480)              // tile diagonal reach = 495
#define WREG    992                      // per-wave accumulator span (16*31+15+480)
#define WSTRIDE 1024

typedef __attribute__((ext_vector_type(16))) __bf16    v16bf;
typedef __attribute__((ext_vector_type(8)))  float     v8f;
typedef __attribute__((ext_vector_type(4)))  unsigned  uv4;
typedef __attribute__((ext_vector_type(8)))  int       iv8;
typedef __attribute__((ext_vector_type(4)))  int       iv4;

// ---------------- workspace layout (bytes) ----------------
#define OFF_ACC     ((size_t)0)
#define SZ_ACC      ((size_t)NCH * NSEG * ACCLEN * 4)     // K partials
#define OFF_RESBF   (OFF_ACC + SZ_ACC)
#define SZ_RESBF    ((size_t)NCH * NSAMP * 2)             // res raw bf16
#define OFF_RREV    (OFF_RESBF + SZ_RESBF)
#define SZ_RREV     ((size_t)NCH * REVPAD * 2)            // reversed+scaled res bf16
#define OFF_IMP     (OFF_RREV + SZ_RREV)
#define SZ_IMP      ((size_t)CPD * ATTACK * 4)            // impulse f32
#define OFF_IMPBF   (OFF_IMP + SZ_IMP)
#define SZ_IMPBF    ((size_t)CPD * ATTACK * 2)            // impulse bf16
#define OFF_BEFORE  (OFF_IMPBF + SZ_IMPBF)
#define SZ_BEFORE   ((size_t)NEV * NRES * FRAMES * 4)
#define OFF_SBUF    (OFF_BEFORE + SZ_BEFORE)
#define SZ_SBUF     ((size_t)8 * ATTACK * 4)
#define OFF_NPART   (OFF_SBUF + SZ_SBUF)
#define SZ_NPART    ((size_t)NCH * 512 * 4)
#define OFF_GPART   (OFF_NPART + SZ_NPART)
#define SZ_GPART    ((size_t)8 * 8 * 4)
#define OFF_SCALE   (OFF_GPART + SZ_GPART)
#define SZ_SCALE    ((size_t)NCH * 4)
#define OFF_APRM    (OFF_SCALE + SZ_SCALE)
#define SZ_APRM     ((size_t)5 * 128 * 4)

// output layout: summed (4*NSAMP) | before (8192) | cs (16*NSAMP)
#define OUT_SUMMED  0
#define OUT_BEFORE  (4 * NSAMP)
#define OUT_CS      (4 * NSAMP + NEV * NRES * FRAMES)

#define USE_TDM (__has_builtin(__builtin_amdgcn_tensor_load_to_lds) && \
                 __has_builtin(__builtin_amdgcn_s_wait_tensorcnt))

#if USE_TDM
// 1-D contiguous bf16 tile load, global -> LDS, via Tensor Data Mover.
// D# packing per CDNA5 ISA 8.3/8.4: group0 {count=1, lds_addr, global_addr, type=2},
// group1 {data_size=2B, tensor_dim0=len, tensor_dim1=1, tile_dim0=len, tile_dim1=1,
//         tensor_dim0_stride=len}.
__device__ __forceinline__ void tdm_load_1d_bf16(unsigned lds_off, const void* gptr, int len)
{
  unsigned long long ga = (unsigned long long)(uintptr_t)gptr;
  uv4 g0;
  g0[0] = 1u;                                  // count = 1, user descriptor
  g0[1] = lds_off;                             // lds_addr
  g0[2] = (unsigned)(ga & 0xffffffffu);        // global_addr[31:0]
  g0[3] = (unsigned)((ga >> 32) & 0x01ffffffu) // global_addr[56:32]
          | 0x80000000u;                       // type = 2 ("image")
  iv8 g1;
  g1[0] = 0x10000;                             // wg_mask=0, data_size=1 (2 bytes)
  g1[1] = (int)((unsigned)(len & 0xffff) << 16);          // tensor_dim0[15:0]
  g1[2] = (int)(((len >> 16) & 0xffff) | (1u << 16));     // dim0[31:16] | tensor_dim1=1
  g1[3] = (int)((unsigned)(len & 0xffff) << 16);          // dim1[31:16]=0 | tile_dim0
  g1[4] = 1;                                   // tile_dim1 = 1, tile_dim2 = 0
  g1[5] = len;                                 // tensor_dim0_stride[31:0]
  g1[6] = 0;                                   // stride0[47:32] | stride1[15:0]
  g1[7] = 0;                                   // stride1[47:16]
  iv4 z = {0, 0, 0, 0};
#if __clang_major__ >= 23
  iv8 z8 = {0, 0, 0, 0, 0, 0, 0, 0};
  __builtin_amdgcn_tensor_load_to_lds(g0, g1, z, z, z8, 0);
#else
  __builtin_amdgcn_tensor_load_to_lds(g0, g1, z, z, 0);
#endif
}
#endif

// =========================================================
// 1) DHO oscillator chain -> raw res (bf16) + norm partials
// =========================================================
__global__ __launch_bounds__(256) void dho_kernel(
    const float* __restrict__ damp, const float* __restrict__ mass,
    const float* __restrict__ tens, const float* __restrict__ disp,
    const float* __restrict__ amp,  const float* __restrict__ infl,
    const float* __restrict__ smix, __bf16* __restrict__ resbf,
    float* __restrict__ npart)
{
  const int ch = blockIdx.y;                 // ch = r*2 + x
  const int r = ch >> 1, x = ch & 1;
  const int n = blockIdx.x * 256 + threadIdx.x;
  const float tval = 10.f * (float)n / (float)(NSAMP - 1);
  float prev = 0.f, o[3];
  #pragma unroll
  for (int j = 0; j < 3; ++j) {
    float s = 0.f;
    #pragma unroll
    for (int osc = 0; osc < 2; ++osc) {
      const int idx = ((j * 2 + osc) * 16 + r) * 2 + x;
      const float ms = 2.f / (1.f + expf(-mass[idx]));     // sigmoid*2
      const float dm = 30.f / (1.f + expf(-damp[idx]));    // sigmoid*30
      const float gma = dm / (2.f * ms);
      float tt = tens[idx];
      if (j > 0) tt += prev * infl[(((j - 1) * 2 + osc) * 16 + r) * 2 + x];
      const float w0sq = exp10f(tt) / ms;
      const float om = sqrtf(w0sq - gma * gma);
      const float x0 = disp[idx];
      const float B = gma * x0 / om;
      const float v = expf(-gma * tval) * (x0 * cosf(om * tval) + B * sinf(om * tval));
      s += amp[idx] * v;
    }
    o[j] = s; prev = s;
  }
  const float e0 = expf(smix[ch * 3 + 0]);
  const float e1 = expf(smix[ch * 3 + 1]);
  const float e2 = expf(smix[ch * 3 + 2]);
  const float res = (e0 * o[0] + e1 * o[1] + e2 * o[2]) / (e0 + e1 + e2);
  resbf[(size_t)ch * NSAMP + n] = (__bf16)res;

  __shared__ float red[256];
  red[threadIdx.x] = res * res;
  __syncthreads();
  #pragma unroll
  for (int s2 = 128; s2 > 0; s2 >>= 1) {
    if (threadIdx.x < s2) red[threadIdx.x] += red[threadIdx.x + s2];
    __syncthreads();
  }
  if (threadIdx.x == 0) npart[ch * 512 + blockIdx.x] = red[0];
}

// double unit_norm folded into one scale: z = x / ((n1+eps)*(n1/(n1+eps)+eps))
__global__ void scale_kernel(const float* __restrict__ npart, float* __restrict__ scalebuf)
{
  const int ch = threadIdx.x;
  if (ch >= NCH) return;
  float s = 0.f;
  for (int b = 0; b < 512; ++b) s += npart[ch * 512 + b];
  const float n1 = sqrtf(s);
  const float d1 = n1 + EPSF;
  const float n2 = n1 / d1;
  scalebuf[ch] = 1.f / (d1 * (n2 + EPSF));
}

// reversed + scaled res, zero-padded: resrev[ch][u] = scale*res[ch][NSAMP-1-u]
__global__ __launch_bounds__(256) void res_rev_kernel(const __bf16* __restrict__ resbf,
                                                      const float* __restrict__ scalebuf,
                                                      __bf16* __restrict__ resrev)
{
  const int ch = blockIdx.y;
  const int u = blockIdx.x * 256 + threadIdx.x;
  if (u >= REVPAD) return;
  const float scale = scalebuf[ch];
  const int sidx = NSAMP - 1 - u;
  float v = 0.f;
  if (sidx >= 0) v = (float)resbf[(size_t)ch * NSAMP + sidx] * scale;
  resrev[(size_t)ch * REVPAD + u] = (__bf16)v;
}

// =========================================================
// 2) attack envelopes: hierarchical dirac == shift; gamma pdf
// =========================================================
__global__ void attack_params_kernel(const float* __restrict__ m_, const float* __restrict__ s_,
                                     const float* __restrict__ a_, const float* __restrict__ pos_,
                                     float* __restrict__ prm)
{
  const int t = threadIdx.x;           // t = c*8 + g
  if (t >= 128) return;
  const float m = fabsf(m_[t]), sd = fabsf(s_[t]), a = fabsf(a_[t]);
  float kr = m / (sd + EPSF);
  const float k = kr * kr;
  const float rate = m / (sd * sd + EPSF);
  const float lco = k * logf(rate + EPSF) - lgammaf(k);
  int p = 0;
  for (int l = 0; l < 11; ++l) {
    const float p0 = pos_[(t * 11 + l) * 2 + 0];
    const float p1 = pos_[(t * 11 + l) * 2 + 1];
    if (p1 > p0) p |= (1 << (10 - l));   // argmax(first-max) bit
  }
  prm[t] = k; prm[128 + t] = rate; prm[256 + t] = lco;
  prm[384 + t] = a; prm[512 + t] = (float)p;
}

__global__ __launch_bounds__(256) void attack_S_kernel(const float* __restrict__ prm,
                                                       float* __restrict__ Sbuf,
                                                       float* __restrict__ gpart)
{
  const int n = blockIdx.x * 256 + threadIdx.x;   // 0..2047
  const float* kk = prm;         const float* rate = prm + 128;
  const float* lco = prm + 256;  const float* aa = prm + 384;
  const float* pp = prm + 512;
  float sq[8];
  #pragma unroll
  for (int g = 0; g < 8; ++g) {
    float s = 0.f;
    for (int c = 0; c < CPD; ++c) {
      const int idx = c * 8 + g;
      const int j = n - (int)pp[idx];
      if (j >= 0) {
        const float t = EPSF + (float)j * ((1.f - EPSF) / 2047.f);  // linspace(EPS,1,2048)
        const float lp = lco[idx] + (kk[idx] - 1.f) * logf(t) - rate[idx] * t;
        s += expf(lp) * aa[idx];
      }
    }
    Sbuf[g * ATTACK + n] = s;
    sq[g] = s * s;
  }
  __shared__ float red[256];
  for (int g = 0; g < 8; ++g) {
    red[threadIdx.x] = sq[g];
    __syncthreads();
    #pragma unroll
    for (int s2 = 128; s2 > 0; s2 >>= 1) {
      if (threadIdx.x < s2) red[threadIdx.x] += red[threadIdx.x + s2];
      __syncthreads();
    }
    if (threadIdx.x == 0) gpart[g * 8 + blockIdx.x] = red[0];
    __syncthreads();
  }
}

__global__ void attack_fin_kernel(const float* __restrict__ Sbuf, const float* __restrict__ gpart,
                                  const float* __restrict__ oamp, const float* __restrict__ noise,
                                  float* __restrict__ impf, __bf16* __restrict__ impbf)
{
  const int n = blockIdx.x * 256 + threadIdx.x;   // < 2048
  float T = 0.f;
  #pragma unroll
  for (int g = 0; g < 8; ++g) {
    float gs = 0.f;
    #pragma unroll
    for (int b = 0; b < 8; ++b) gs += gpart[g * 8 + b];
    T += Sbuf[g * ATTACK + n] / (sqrtf(gs) + EPSF);
  }
  for (int c = 0; c < CPD; ++c) {
    const float v = oamp[c] * T * noise[c * ATTACK + n];
    impf[c * ATTACK + n] = v;
    impbf[c * ATTACK + n] = (__bf16)v;
  }
}

// =========================================================
// 3) router einsum (exact f32) -> `before` output + ws copy
// =========================================================
__global__ void router_kernel(const float* __restrict__ control, const float* __restrict__ router,
                              float* __restrict__ beforews, float* __restrict__ out_before)
{
  const int t = blockIdx.x * 256 + threadIdx.x;     // 8192 = (e,r,f)
  const int e = t >> 11, rr = (t >> 7) & 15, f = t & 127;
  float acc = 0.f;
  #pragma unroll
  for (int c = 0; c < 16; ++c)
    acc += control[(e * 16 + c) * 128 + f] * router[c * 16 + rr];
  beforews[t] = acc;
  out_before[t] = acc;
}

// =========================================================
// 4) K = impulse (x) res  via bf16 WMMA skewed tiles.
//    D[i,j] (tile n_base) accumulates out[n_base+i+32j] over
//    taps m = 512p + 32j + k, p chained through the C operand.
//    LDS staged by the Tensor Data Mover when available.
// =========================================================
__global__ __launch_bounds__(256) void conv_wmma_kernel(
    const __bf16* __restrict__ resrev, const __bf16* __restrict__ impbf,
    float* __restrict__ accbuf)
{
  __shared__ __align__(16) __bf16 R[RLEN + 2];   // reversed scaled res (+pad)
  __shared__ __align__(16) __bf16 H[ATTACK];     // impulse taps
  __shared__ unsigned RS[RLEN / 2];              // shifted packed pairs (2q+1, 2q+2)
  __shared__ float WA[8][WSTRIDE];               // per-wave private accumulators
  const int ch = blockIdx.y;                     // ch = 2r + x
  const int r = ch >> 1;
  const int seg = blockIdx.x;
  const int seg0 = seg * SEG;
  const int tid = threadIdx.x;
  const int lane = tid & 31, wave = tid >> 5;
  const int ustart = NSAMP - seg0 - SEG;         // resrev offset of R[0]

  // phase 1: zero accumulators + stage R/H into LDS
  for (int i2 = tid; i2 < 8 * WSTRIDE; i2 += 256) (&WA[0][0])[i2] = 0.f;
  if (tid == 0) { R[RLEN] = (__bf16)0.f; R[RLEN + 1] = (__bf16)0.f; }
#if USE_TDM
  if (wave == 0) {
    tdm_load_1d_bf16((unsigned)(uintptr_t)&R[0],
                     resrev + (size_t)ch * REVPAD + ustart, RLEN);
    tdm_load_1d_bf16((unsigned)(uintptr_t)&H[0],
                     impbf + (size_t)r * ATTACK, ATTACK);
    __builtin_amdgcn_s_wait_tensorcnt(0);
  }
#else
  for (int t = tid; t < RLEN; t += 256)
    R[t] = resrev[(size_t)ch * REVPAD + ustart + t];
  for (int m = tid; m < ATTACK; m += 256)
    H[m] = impbf[(size_t)r * ATTACK + m];
#endif
  __syncthreads();

  // phase 2: build shifted packed-pair shadow for aligned b32 A-operand loads
  {
    const unsigned short* Ru = (const unsigned short*)R;
    for (int q = tid; q < RLEN / 2; q += 256)
      RS[q] = (unsigned)Ru[2 * q + 1] | ((unsigned)Ru[2 * q + 2] << 16);
  }
  __syncthreads();

  const unsigned* RW = (const unsigned*)R;       // packed pairs (2q, 2q+1)
  const uv4* H4 = (const uv4*)H;
  const int hi = lane >> 4, li = lane & 15;
  // 16-bit A 16x32 layout: lanes<16 hold K {0..7,16..23}, lanes>=16 hold K {8..15,24..31}
  const int k0a = hi ? 8 : 0, k0b = hi ? 24 : 16;
  // 16-bit B 32x16 layout: lanes<16 hold K {0..7,16..23}, lanes>=16 hold K {8..15,24..31}, col j = lane%16
  const int bofs = 32 * li + (hi ? 8 : 0);
  float* wacc = WA[wave];

  union AV { v16bf v; unsigned w[8]; };
  union BV { v16bf v; uv4 u[2]; };

  for (int Tl = 0; Tl < 32; ++Tl) {              // 32 contiguous tiles per wave
    const int T = wave * 32 + Tl;
    const int tb = SEG - 1 - 16 * T;             // smax - n_base
    v8f c = {};
    #pragma unroll
    for (int p = 0; p < 4; ++p) {
      const int ub = tb + 512 * p - li;          // A[i,k] = R[ub + k]
      AV a; BV b;
      const int b1 = ub + k0a, b2 = ub + k0b;
      #pragma unroll
      for (int q = 0; q < 4; ++q) {              // aligned b32 pair loads
        const int i1 = b1 + 2 * q;
        a.w[q]     = (i1 & 1) ? RS[i1 >> 1] : RW[i1 >> 1];
        const int i2 = b2 + 2 * q;
        a.w[4 + q] = (i2 & 1) ? RS[i2 >> 1] : RW[i2 >> 1];
      }
      const int bb = 512 * p + bofs;             // B[k,j] = H[512p + 32j + k]
      b.u[0] = H4[bb >> 3];
      b.u[1] = H4[(bb >> 3) + 2];
      c = __builtin_amdgcn_wmma_f32_16x16x32_bf16(false, a.v, false, b.v,
                                                  (short)0, c, false, false);
    }
    // D layout: lane -> col j=li; VGPR v -> row i = v + 8*hi.  out += D at 16T+i+32j
    const int pbase = 16 * Tl + 32 * li + 8 * hi;
    #pragma unroll
    for (int v = 0; v < 8; ++v) wacc[pbase + v] += c[v];
  }
  __syncthreads();

  // merge the 8 wave regions (deterministic, no atomics) into segment partials
  float* out = accbuf + ((size_t)ch * NSEG + seg) * ACCLEN;
  for (int idx = tid; idx < ACCLEN; idx += 256) {
    const int w0 = idx >> 9;
    float v = 0.f;
    if (w0 < 8)  { const int l = idx - (w0 << 9);       if (l < WREG) v += WA[w0][l]; }
    if (w0 >= 1) { const int l = idx - ((w0 - 1) << 9); if (l < WREG) v += WA[w0 - 1][l]; }
    out[idx] = v;
  }
}

// =========================================================
// 5) frame mix + deformation + layer mix + tanh -> outputs
// =========================================================
__device__ __forceinline__ float kv_load(const float* __restrict__ A, int m)
{
  const int s = m >> 12, l = m & 4095;
  float v = A[(size_t)s * ACCLEN + l];
  if (l < 480 && s > 0) v += A[(size_t)(s - 1) * ACCLEN + 4096 + l];  // segment halo
  return v;
}

__global__ __launch_bounds__(256) void final_kernel(
    const float* __restrict__ beforew, const float* __restrict__ impf,
    const float* __restrict__ accbuf, const float* __restrict__ deform,
    const float* __restrict__ lmix, const float* __restrict__ gains,
    float* __restrict__ out_summed, float* __restrict__ out_cs)
{
  __shared__ float Bs[NEV * NRES * FRAMES];
  const int tid = threadIdx.x;
  for (int i = tid; i < NEV * NRES * FRAMES; i += 256) Bs[i] = beforew[i];
  __syncthreads();

  const int n = blockIdx.x * 256 + tid;
  // deformation softmax (base adds 1 to channel 0) + linear interp 128 -> NSAMP
  float src = ((float)n + 0.5f) * (128.f / (float)NSAMP) - 0.5f;
  src = fminf(fmaxf(src, 0.f), 127.f);
  const int lo = (int)src;
  int hi = lo + 1; if (hi > 127) hi = 127;
  const float w = src - (float)lo;
  float dl0, dl1, dh0, dh1;
  { const float a0 = 1.f + deform[lo], a1 = deform[128 + lo];
    const float mx = fmaxf(a0, a1);
    const float e0 = expf(a0 - mx), e1 = expf(a1 - mx), s = e0 + e1;
    dl0 = e0 / s; dl1 = e1 / s; }
  { const float a0 = 1.f + deform[hi], a1 = deform[128 + hi];
    const float mx = fmaxf(a0, a1);
    const float e0 = expf(a0 - mx), e1 = expf(a1 - mx), s = e0 + e1;
    dh0 = e0 / s; dh1 = e1 / s; }
  const float d0 = dl0 * (1.f - w) + dh0 * w;
  const float d1 = dl1 * (1.f - w) + dh1 * w;

  const int f0 = n >> 10, o0 = n & 1023;
  float summed[4] = {0.f, 0.f, 0.f, 0.f};

  for (int rr = 0; rr < 16; ++rr) {
    const float m0 = lmix[rr * 2], m1 = lmix[rr * 2 + 1];
    const float mx = fmaxf(m0, m1);
    const float e0 = expf(m0 - mx), e1 = expf(m1 - mx), es = e0 + e1;
    const float w0 = e0 / es, w1 = e1 / es;
    const float imp0 = impf[rr * ATTACK + o0];
    const float imp1 = impf[rr * ATTACK + o0 + 1024];
    const float* A0 = accbuf + (size_t)(2 * rr) * NSEG * ACCLEN;
    const float* A1 = A0 + (size_t)NSEG * ACCLEN;
    __builtin_prefetch(&A0[(size_t)(n >> 12) * ACCLEN + (n & 4095)], 0, 1);
    __builtin_prefetch(&A1[(size_t)(n >> 12) * ACCLEN + (n & 4095)], 0, 1);
    float c0[4] = {0.f, 0.f, 0.f, 0.f}, c1[4] = {0.f, 0.f, 0.f, 0.f};
    for (int f = 0; f <= f0; ++f) {
      const int m = n - (f << 10);
      const float k0 = kv_load(A0, m);
      const float k1 = kv_load(A1, m);
      #pragma unroll
      for (int e = 0; e < 4; ++e) {
        const float bf = Bs[(e * 16 + rr) * 128 + f];
        c0[e] += bf * k0; c1[e] += bf * k1;
      }
    }
    float cs = 0.f;
    const int q = rr & 3, rh = rr >> 2;
    #pragma unroll
    for (int e = 0; e < 4; ++e) {
      float rs = Bs[(e * 16 + rr) * 128 + f0] * imp0;             // frame f0 tap
      if (f0 >= 1) rs += Bs[(e * 16 + rr) * 128 + f0 - 1] * imp1; // frame f0-1 tap
      cs += rs;
      const float cx = d0 * c0[e] + d1 * c1[e];
      const float mixed = w0 * rs + w1 * cx;
      const float g = fabsf(gains[4 * e + rh]);                   // gains[rho=4e+r//4]
      summed[q] += tanhf(mixed * g);
    }
    out_cs[(size_t)rr * NSAMP + n] = cs;
  }
  #pragma unroll
  for (int q = 0; q < 4; ++q) out_summed[(size_t)q * NSAMP + n] = summed[q];
}

// =========================================================
extern "C" void kernel_launch(void* const* d_in, const int* in_sizes, int n_in,
                              void* d_out, int out_size, void* d_ws, size_t ws_size,
                              hipStream_t stream)
{
  (void)in_sizes; (void)n_in; (void)out_size; (void)ws_size;
  const float* control  = (const float*)d_in[0];
  const float* deform   = (const float*)d_in[1];
  const float* ae_means = (const float*)d_in[2];
  const float* ae_stds  = (const float*)d_in[3];
  const float* ae_amps  = (const float*)d_in[4];
  const float* ae_pos   = (const float*)d_in[5];
  const float* ae_oamp  = (const float*)d_in[6];
  const float* dho_damp = (const float*)d_in[7];
  const float* dho_mass = (const float*)d_in[8];
  const float* dho_tens = (const float*)d_in[9];
  const float* dho_disp = (const float*)d_in[10];
  const float* dho_amp  = (const float*)d_in[11];
  const float* infl     = (const float*)d_in[12];
  const float* smix     = (const float*)d_in[13];
  const float* router   = (const float*)d_in[14];
  const float* lmix     = (const float*)d_in[15];
  const float* gains    = (const float*)d_in[16];
  const float* noise    = (const float*)d_in[17];

  char* ws = (char*)d_ws;
  float*  accbuf   = (float*)(ws + OFF_ACC);
  __bf16* resbf    = (__bf16*)(ws + OFF_RESBF);
  __bf16* resrev   = (__bf16*)(ws + OFF_RREV);
  float*  impf     = (float*)(ws + OFF_IMP);
  __bf16* impbf    = (__bf16*)(ws + OFF_IMPBF);
  float*  beforews = (float*)(ws + OFF_BEFORE);
  float*  Sbuf     = (float*)(ws + OFF_SBUF);
  float*  npart    = (float*)(ws + OFF_NPART);
  float*  gpart    = (float*)(ws + OFF_GPART);
  float*  scalebuf = (float*)(ws + OFF_SCALE);
  float*  aprm     = (float*)(ws + OFF_APRM);

  float* out = (float*)d_out;

  dho_kernel<<<dim3(NSAMP / 256, NCH), 256, 0, stream>>>(
      dho_damp, dho_mass, dho_tens, dho_disp, dho_amp, infl, smix, resbf, npart);
  scale_kernel<<<1, 32, 0, stream>>>(npart, scalebuf);
  res_rev_kernel<<<dim3((REVPAD + 255) / 256, NCH), 256, 0, stream>>>(
      resbf, scalebuf, resrev);
  attack_params_kernel<<<1, 128, 0, stream>>>(ae_means, ae_stds, ae_amps, ae_pos, aprm);
  attack_S_kernel<<<8, 256, 0, stream>>>(aprm, Sbuf, gpart);
  attack_fin_kernel<<<8, 256, 0, stream>>>(Sbuf, gpart, ae_oamp, noise, impf, impbf);
  router_kernel<<<32, 256, 0, stream>>>(control, router, beforews, out + OUT_BEFORE);
  conv_wmma_kernel<<<dim3(NSEG, NCH), 256, 0, stream>>>(resrev, impbf, accbuf);
  final_kernel<<<NSAMP / 256, 256, 0, stream>>>(
      beforews, impf, accbuf, deform, lmix, gains, out + OUT_SUMMED, out + OUT_CS);
}
